// torchLSTM_cell_open_loop_44598940402205
// MI455X (gfx1250) — matrix operand order repaired and checked by reading, not compiled, not run on hardware
//
#include <hip/hip_runtime.h>

// ---------------------------------------------------------------------------
// Persistent single-workgroup LSTM scan for MI455X (gfx1250, wave32, WMMA).
// Latency-bound sequential recurrence (65536 dependent steps): minimize the
// per-step critical path. 5 waves, weights pinned in VGPRs as WMMA B-frags,
// state in LDS, gates via v_wmma_f32_16x16x32_f16 (4 per wave per step),
// x_{t+1} register-prefetched during step t, activations on v_tanh_f32,
// WGP-scope global_prefetch_b8 keeps x rows resident in the WGP cache.
// ---------------------------------------------------------------------------

typedef __attribute__((ext_vector_type(16))) _Float16 v16h;
typedef __attribute__((ext_vector_type(8)))  _Float16 v8h;
typedef __attribute__((ext_vector_type(8)))  float    v8f;

#define NT       65536
#define NG       28            // ngrid
#define NXF      19            // nx
#define HID      20            // H
#define NGATE    80            // 4*H
#define KDIM     64            // K = 40 padded to 2 chunks of 32
#define ZPITCH   72            // Z row pitch (16B-aligned rows)
#define NTHREADS 160           // 5 waves of 32
#define NXTOT    (NG * NXF)    // 532 floats of x per step
#define NPAIR    (NXTOT / 2)   // 266 float2 loads per step
#define NCELL    (NG * HID)    // 560 LSTM cells

// gfx1250 hardware tanh (TRANS32). v_nop x2 covers the documented TRANS
// result-use hazard (hardware does not interlock co-executing multicycle ops).
__device__ __forceinline__ float tanh_hw(float x) {
    float r;
    asm("v_tanh_f32 %0, %1\n\tv_nop\n\tv_nop" : "=v"(r) : "v"(x));
    return r;
}
__device__ __forceinline__ float sig_hw(float x) {
    return fmaf(tanh_hw(0.5f * x), 0.5f, 0.5f);   // sigmoid(x)=0.5+0.5*tanh(x/2)
}

// Prefetch into ALL cache levels (default scope 0 = WGP; the builtin only
// reaches SE/SYS scope which skips the WGP cache).
__device__ __forceinline__ void prefetch_wgp(const void* p) {
    asm volatile("global_prefetch_b8 %0, off" :: "v"(p));
}

__global__ __launch_bounds__(NTHREADS, 1)
void lstm_scan_kernel(const float* __restrict__ x,
                      const float* __restrict__ W_ih,
                      const float* __restrict__ W_hh,
                      const float* __restrict__ b_ih,
                      const float* __restrict__ b_hh,
                      const float* __restrict__ W_out,
                      const float* __restrict__ b_out,
                      float* __restrict__ out)
{
    // Z = [x_t | prev | h | 0-pad] f16; rows 28..31 and cols 40..63 stay 0.
    __shared__ __align__(16) _Float16 Zsh[32][ZPITCH];
    __shared__ __align__(16) _Float16 Ush[KDIM][NGATE];   // init staging only
    __shared__ __align__(16) float    Gsh[32][NGATE];     // gate pre-activations
    __shared__ __align__(16) float    Hsh[NG][HID];       // f32 h for output dot
    __shared__ float bias_sh[NGATE];
    __shared__ float wout_sh[HID];

    const int tid  = threadIdx.x;
    const int lane = tid & 31;
    const int wave = tid >> 5;            // 0..4 -> owns N-tile [16w, 16w+16)

    // ---------------- one-time init ----------------
    for (int i = tid; i < 32 * ZPITCH; i += NTHREADS)
        (&Zsh[0][0])[i] = (_Float16)0.0f;
    for (int i = tid; i < NG * HID; i += NTHREADS)
        (&Hsh[0][0])[i] = 0.0f;
    // Fused weight U[k][n]: k<20 -> W_ih[n][k]; 20<=k<40 -> W_hh[n][k-20]
    for (int i = tid; i < KDIM * NGATE; i += NTHREADS) {
        int k = i / NGATE, n = i % NGATE;
        float v = 0.0f;
        if (k < HID)          v = W_ih[n * HID + k];
        else if (k < 2 * HID) v = W_hh[n * HID + (k - HID)];
        Ush[k][n] = (_Float16)v;
    }
    for (int i = tid; i < NGATE; i += NTHREADS) bias_sh[i] = b_ih[i] + b_hh[i];
    for (int i = tid; i < HID;   i += NTHREADS) wout_sh[i] = W_out[i];
    __syncthreads();

    // Persistent B fragments (32x16 f16): lane = N (lane&15),
    // element e -> K = 16*(lane>=16) + e.
    const int n0  = wave * 16;
    const int bn  = lane & 15;
    const int bkg = (lane >> 4) * 16;
    v16h bf0, bf1;
    for (int e = 0; e < 16; ++e) {
        bf0[e] = Ush[ 0 + bkg + e][n0 + bn];
        bf1[e] = Ush[32 + bkg + e][n0 + bn];
    }
    v8f biasvec;
    {
        const float bv = bias_sh[n0 + bn];     // C layout: N = lane&15
        for (int e = 0; e < 8; ++e) biasvec[e] = bv;
    }
    const float  bo  = b_out[0];
    const float4 wr0 = *(const float4*)&wout_sh[0];
    const float4 wr1 = *(const float4*)&wout_sh[4];
    const float4 wr2 = *(const float4*)&wout_sh[8];
    const float4 wr3 = *(const float4*)&wout_sh[12];
    const float4 wr4 = *(const float4*)&wout_sh[16];

    // A fragment (16x32 f16): lane = M (lane&15); elements 0..15 -> K=ka..ka+15,
    // ka = 16*(lane>=16). Two 16B LDS loads each.
    const int arow_l = lane & 15;
    const int aka    = (lane >> 4) * 16;
    auto loadA = [&](int mtile, int kbase) -> v16h {
        const int row = mtile * 16 + arow_l;
        v8h lo = *(const v8h*)&Zsh[row][kbase + aka];
        v8h hi = *(const v8h*)&Zsh[row][kbase + aka + 8];
        v16h a;
        for (int e = 0; e < 8; ++e) { a[e] = lo[e]; a[8 + e] = hi[e]; }
        return a;
    };

    // C-tile store pointers (fixed): M = vgpr + 8*(lane>=16), N = lane&15.
    const int cmb = (lane >> 4) * 8;
    float* g0p = &Gsh[ 0 + cmb][n0 + bn];
    float* g1p = &Gsh[16 + cmb][n0 + bn];

    // -------- precomputed x-staging addressing (fixed across t) --------
    // Pair 0 = tid (always valid). Pair 1 clamped: threads >=106 redundantly
    // handle pair 265 (same value to same LDS address -> benign), so the
    // steady-state loop is branch-free here.
    const int p1  = (tid + NTHREADS < NPAIR) ? (tid + NTHREADS) : (NPAIR - 1);
    const int i0  = 2 * tid;
    const int i1  = 2 * p1;
    _Float16* xz0a = &Zsh[(i0    ) / NXF][(i0    ) % NXF];
    _Float16* xz0b = &Zsh[(i0 + 1) / NXF][(i0 + 1) % NXF];
    _Float16* xz1a = &Zsh[(i1    ) / NXF][(i1    ) % NXF];
    _Float16* xz1b = &Zsh[(i1 + 1) / NXF][(i1 + 1) % NXF];

    // -------- precomputed elementwise addressing + register c-state --------
    // Slots 0..2 are valid for every thread (tid+320 < 560); only slot 3
    // needs a guard (tid < 80).
    float*    gptr[4];
    float*    hptr[4];
    _Float16* zptr[4];
    float     creg[4];
    const bool c3v = (tid + 3 * NTHREADS) < NCELL;
#pragma unroll
    for (int k = 0; k < 4; ++k) {
        int idx = tid + k * NTHREADS;
        int ii  = (idx < NCELL) ? idx : 0;
        int g = ii / HID, j = ii % HID;
        gptr[k] = &Gsh[g][j];
        hptr[k] = &Hsh[g][j];
        zptr[k] = &Zsh[g][HID + j];
        creg[k] = 0.0f;
    }

    // One LSTM cell update (gate order i,f,g,o); c stays in a register.
    auto doCell = [&](int k) {
        float gi = gptr[k][0];
        float gf = gptr[k][HID];
        float gg = gptr[k][2 * HID];
        float go = gptr[k][3 * HID];
        float c  = sig_hw(gf) * creg[k] + sig_hw(gi) * tanh_hw(gg);
        float h  = sig_hw(go) * tanh_hw(c);
        creg[k]  = c;
        *hptr[k] = h;
        *zptr[k] = (_Float16)h;                  // Z cols 20..39
    };

    // preload x(0) into registers
    float2 xr0 = *(const float2*)(x + i0);
    float2 xr1 = *(const float2*)(x + i1);

    // ---------------- sequential scan ----------------
    for (int t = 0; t < NT; ++t) {
        // P1: commit x_t (prefetched last step) into Z cols 0..18.
        *xz0a = (_Float16)xr0.x;
        *xz0b = (_Float16)xr0.y;
        *xz1a = (_Float16)xr1.x;
        *xz1b = (_Float16)xr1.y;
        __syncthreads();

        // P2: gates = Z @ U + bias. Batch all 8 ds_load_b128, then 4 WMMAs.
        v16h a00 = loadA(0, 0);
        v16h a01 = loadA(0, 32);
        v16h a10 = loadA(1, 0);
        v16h a11 = loadA(1, 32);
        asm("" :: "v"(a00), "v"(a01), "v"(a10), "v"(a11));
        v8f acc0 = biasvec, acc1 = biasvec;
        acc0 = __builtin_amdgcn_wmma_f32_16x16x32_f16(false, a00, false, bf0,
                                                      (short)0, acc0, false, false);
        acc1 = __builtin_amdgcn_wmma_f32_16x16x32_f16(false, a10, false, bf0,
                                                      (short)0, acc1, false, false);
        acc0 = __builtin_amdgcn_wmma_f32_16x16x32_f16(false, a01, false, bf1,
                                                      (short)0, acc0, false, false);
        acc1 = __builtin_amdgcn_wmma_f32_16x16x32_f16(false, a11, false, bf1,
                                                      (short)0, acc1, false, false);
#pragma unroll
        for (int v = 0; v < 8; ++v) {
            g0p[v * NGATE] = acc0[v];
            g1p[v * NGATE] = acc1[v];
        }

        // register-prefetch x_{t+1} (clamped at the end: harmless reload of
        // the last row); thanks to the WGP-scope prefetch below these are
        // L0 hits, so the pre-barrier loadcnt wait is cheap.
        {
            const int tn = (t + 1 < NT) ? (t + 1) : t;
            const float* xn = x + (size_t)tn * NXTOT;
            xr0 = *(const float2*)(xn + i0);
            xr1 = *(const float2*)(xn + i1);
        }
        if (wave == 0) {
            const int tp = (t + 8 < NT) ? (t + 8) : (NT - 1);
            prefetch_wgp(x + (size_t)tp * NXTOT + lane * 32);
        }
        __syncthreads();

        // P3: LSTM cells; slots 0..2 unconditional, slot 3 guarded once.
        doCell(0);
        doCell(1);
        doCell(2);
        if (c3v) doCell(3);
        __syncthreads();

        // P4: out = h @ W_out^T + b_out; detached feedback -> Z col 19.
        if (tid < NG) {
            const float4* hv = (const float4*)&Hsh[tid][0];
            float4 h0 = hv[0], h1 = hv[1], h2 = hv[2], h3 = hv[3], h4 = hv[4];
            float acc = bo;
            acc += h0.x * wr0.x + h0.y * wr0.y + h0.z * wr0.z + h0.w * wr0.w;
            acc += h1.x * wr1.x + h1.y * wr1.y + h1.z * wr1.z + h1.w * wr1.w;
            acc += h2.x * wr2.x + h2.y * wr2.y + h2.z * wr2.z + h2.w * wr2.w;
            acc += h3.x * wr3.x + h3.y * wr3.y + h3.z * wr3.z + h3.w * wr3.w;
            acc += h4.x * wr4.x + h4.y * wr4.y + h4.z * wr4.z + h4.w * wr4.w;
            out[(size_t)t * NG + tid] = acc;
            Zsh[tid][NXF] = (_Float16)acc;
        }
        // next iteration's barrier (after P1) orders P4 writes vs P2 reads
    }
}

extern "C" void kernel_launch(void* const* d_in, const int* in_sizes, int n_in,
                              void* d_out, int out_size, void* d_ws, size_t ws_size,
                              hipStream_t stream) {
    (void)in_sizes; (void)n_in; (void)out_size; (void)d_ws; (void)ws_size;
    lstm_scan_kernel<<<1, NTHREADS, 0, stream>>>(
        (const float*)d_in[0],   // x        (65536, 28, 19)
        (const float*)d_in[1],   // W_ih     (80, 20)
        (const float*)d_in[2],   // W_hh     (80, 20)
        (const float*)d_in[3],   // b_ih     (80,)
        (const float*)d_in[4],   // b_hh     (80,)
        (const float*)d_in[5],   // W_out    (1, 20)
        (const float*)d_in[6],   // b_out    (1,)
        (float*)d_out);          // out      (65536, 28, 1)
}
